// FusedQKVMatmul_13932873908401
// MI455X (gfx1250) — matrix operand order repaired and checked by reading, not compiled, not run on hardware
//
#include <hip/hip_runtime.h>

// ---------------------------------------------------------------------------
// Fused QKV projection for MI455X (gfx1250, wave32, WMMA, async-to-LDS).
// GEMM: M=8192 (b*s), K=2048, N=6144 (q|k|v), fp32 in/out.
// Phase 1: one-time fp32 -> (hi,lo) bf16 split into workspace.
// Phase 2: bf16x3 GEMM; tiles staged with GLOBAL_LOAD_ASYNC_TO_LDS_B128
//          (ASYNCcnt) into a double-buffered LDS, consumed by
//          v_wmma_f32_16x16x32_bf16.
// ---------------------------------------------------------------------------

#define HIDDEN   2048
#define SEQ      4096
#define BSZ      2
#define NHEADS   16
#define HDIM     128
#define MTOT     (BSZ * SEQ)        // 8192
#define NTOT     (3 * HIDDEN)       // 6144
#define OUT_PER  ((size_t)BSZ * SEQ * HIDDEN)

#define BM 256
#define BN 128
#define BK 32
#define LDA 40                      // ushorts per LDS row (32 data + 8 pad = 80B)

// LDS buffer layout (bytes, per buffer): Ahi | Alo | Bhi | Blo
#define L_AHI 0
#define L_ALO (BM * LDA * 2)                 // 20480
#define L_BHI (2 * BM * LDA * 2)             // 40960
#define L_BLO (2 * BM * LDA * 2 + BN * LDA * 2)  // 51200
#define L_BUF (2 * (BM + BN) * LDA * 2)      // 61440 bytes per buffer

typedef __attribute__((ext_vector_type(16))) __bf16    bf16x16;
typedef __attribute__((ext_vector_type(8)))  float     f32x8;
typedef __attribute__((ext_vector_type(4)))  unsigned  u32x4;
typedef __attribute__((ext_vector_type(2)))  unsigned  u32x2;

union FragCvt { u32x4 u[2]; bf16x16 f; };

__device__ __forceinline__ unsigned bf16rne(float x) {
    union { float f; unsigned u; } c; c.f = x;
    return (c.u + 0x7FFFu + ((c.u >> 16) & 1u)) >> 16;
}
__device__ __forceinline__ float bf16tof(unsigned h) {
    union { unsigned u; float f; } c; c.u = h << 16;
    return c.f;
}

__device__ __forceinline__ bf16x16 load_frag(const unsigned short* p, int second) {
    FragCvt c;
    c.u[0] = *(const u32x4*)(p);
    c.u[1] = *(const u32x4*)(p + second);
    return c.f;
}

// CDNA5 async copy: global -> LDS, 16B per lane, tracked by ASYNCcnt.
__device__ __forceinline__ void async_b128(unsigned ldsByteAddr, const void* sbase,
                                           unsigned voffBytes) {
    asm volatile("global_load_async_to_lds_b128 %0, %1, %2"
                 :: "v"(ldsByteAddr), "v"(voffBytes), "s"(sbase) : "memory");
}
__device__ __forceinline__ void wait_async0() {
    asm volatile("s_wait_asynccnt 0x0" ::: "memory");
}

// ---------------------------------------------------------------------------
// Phase 1: split n8*8 fp32 values into hi/lo bf16 arrays (16B stores).
// ---------------------------------------------------------------------------
__global__ __launch_bounds__(256)
void split_bf16_kernel(const float* __restrict__ src,
                       unsigned short* __restrict__ hi,
                       unsigned short* __restrict__ lo, int n8) {
    int t = blockIdx.x * 256 + threadIdx.x;
    if (t >= n8) return;
    float4 v0 = ((const float4*)src)[(size_t)t * 2];
    float4 v1 = ((const float4*)src)[(size_t)t * 2 + 1];
    const float* v[2] = { (const float*)&v0, (const float*)&v1 };
    unsigned h[8], l[8];
#pragma unroll
    for (int q = 0; q < 2; ++q)
#pragma unroll
        for (int e = 0; e < 4; ++e) {
            int k = q * 4 + e;
            h[k] = bf16rne(v[q][e]);
            l[k] = bf16rne(v[q][e] - bf16tof(h[k]));
        }
    u32x4 hv = { h[0] | (h[1] << 16), h[2] | (h[3] << 16),
                 h[4] | (h[5] << 16), h[6] | (h[7] << 16) };
    u32x4 lv = { l[0] | (l[1] << 16), l[2] | (l[3] << 16),
                 l[4] | (l[5] << 16), l[6] | (l[7] << 16) };
    ((u32x4*)hi)[t] = hv;
    ((u32x4*)lo)[t] = lv;
}

// ---------------------------------------------------------------------------
// Phase 2: 256x128-tile bf16x3 GEMM with async LDS staging.
// 512 threads = 16 waves, wave grid 4(M) x 4(N), wave tile 64x32.
// ---------------------------------------------------------------------------
__global__ __launch_bounds__(512)
void fused_qkv_gemm_async(const unsigned short* __restrict__ Ahi,
                          const unsigned short* __restrict__ Alo,
                          const unsigned short* __restrict__ Whi,
                          const unsigned short* __restrict__ Wlo,
                          float* __restrict__ out) {
    extern __shared__ unsigned short smem[];   // 2 * L_BUF bytes

    const int tid  = threadIdx.x;
    const int lane = tid & 31;
    const int wave = tid >> 5;          // 16 waves
    const int wm   = wave & 3;          // 0..3 -> 64-row slab
    const int wn   = wave >> 2;         // 0..3 -> 32-col slab
    const int l16  = lane & 15;
    const int ksel = lane >> 4;

    const int m0 = blockIdx.x * BM;
    const int n0 = blockIdx.y * BN;
    const int which = n0 >> 11;
    const int o0 = n0 & (HIDDEN - 1);

    // Global row origins (bf16, row stride HIDDEN => 4096 bytes).
    const unsigned short* aHi = Ahi + (size_t)m0 * HIDDEN;
    const unsigned short* aLo = Alo + (size_t)m0 * HIDDEN;
    const unsigned short* bHi = Whi + (size_t)n0 * HIDDEN;
    const unsigned short* bLo = Wlo + (size_t)n0 * HIDDEN;

    const unsigned ldsBase = (unsigned)(uintptr_t)smem;   // LDS byte offset
    const int rA0 = tid >> 2;            // 0..127
    const int rA1 = rA0 + 128;           // 128..255
    const int c16 = (tid & 3) * 16;      // 16B chunk within 64B row

    // Per-thread constant pieces of the addresses.
    const unsigned gA0 = (unsigned)rA0 * (HIDDEN * 2) + c16;
    const unsigned gA1 = (unsigned)rA1 * (HIDDEN * 2) + c16;
    const unsigned lA0 = (unsigned)rA0 * (LDA * 2) + c16;
    const unsigned lA1 = (unsigned)rA1 * (LDA * 2) + c16;

    f32x8 acc[4][2] = {};

    // Prologue: stage k-block 0 into buffer 0.
    {
        unsigned nb = ldsBase;
        async_b128(nb + L_AHI + lA0, aHi, gA0);
        async_b128(nb + L_AHI + lA1, aHi, gA1);
        async_b128(nb + L_ALO + lA0, aLo, gA0);
        async_b128(nb + L_ALO + lA1, aLo, gA1);
        async_b128(nb + L_BHI + lA0, bHi, gA0);
        async_b128(nb + L_BLO + lA0, bLo, gA0);
    }

    int cur = 0;
    for (int kb = 0; kb < HIDDEN; kb += BK) {
        wait_async0();          // my async writes into buf[cur] done
        __syncthreads();        // everyone's writes visible / reads of other buf done

        const bool more = (kb + BK) < HIDDEN;
        if (more) {             // stage next k-block into the other buffer
            unsigned ko = (unsigned)(kb + BK) * 2u;
            unsigned nb = ldsBase + (unsigned)(cur ^ 1) * L_BUF;
            async_b128(nb + L_AHI + lA0, aHi, gA0 + ko);
            async_b128(nb + L_AHI + lA1, aHi, gA1 + ko);
            async_b128(nb + L_ALO + lA0, aLo, gA0 + ko);
            async_b128(nb + L_ALO + lA1, aLo, gA1 + ko);
            async_b128(nb + L_BHI + lA0, bHi, gA0 + ko);
            async_b128(nb + L_BLO + lA0, bLo, gA0 + ko);
        }

        const unsigned short* sb = smem + cur * (L_BUF / 2);   // ushort units
        const unsigned short* sAhi = sb;
        const unsigned short* sAlo = sb + L_ALO / 2;
        const unsigned short* sBhi = sb + L_BHI / 2;
        const unsigned short* sBlo = sb + L_BLO / 2;

        bf16x16 bh[2], bl[2];
#pragma unroll
        for (int j = 0; j < 2; ++j) {
            unsigned off = (unsigned)(wn * 32 + j * 16 + l16) * LDA + (unsigned)ksel * 8u;
            bh[j] = load_frag(sBhi + off, 16);
            bl[j] = load_frag(sBlo + off, 16);
        }
#pragma unroll
        for (int i = 0; i < 4; ++i) {
            unsigned off = (unsigned)(wm * 64 + i * 16 + l16) * LDA + (unsigned)ksel * 8u;
            bf16x16 ah = load_frag(sAhi + off, 16);
            bf16x16 al = load_frag(sAlo + off, 16);
#pragma unroll
            for (int j = 0; j < 2; ++j) {
                acc[i][j] = __builtin_amdgcn_wmma_f32_16x16x32_bf16(
                    false, ah, false, bh[j], (short)0, acc[i][j], false, false);
                acc[i][j] = __builtin_amdgcn_wmma_f32_16x16x32_bf16(
                    false, ah, false, bl[j], (short)0, acc[i][j], false, false);
                acc[i][j] = __builtin_amdgcn_wmma_f32_16x16x32_bf16(
                    false, al, false, bh[j], (short)0, acc[i][j], false, false);
            }
        }
        cur ^= 1;
    }

    float* __restrict__ outW = out + (size_t)which * OUT_PER;
#pragma unroll
    for (int i = 0; i < 4; ++i) {
#pragma unroll
        for (int j = 0; j < 2; ++j) {
            const int oc   = o0 + wn * 32 + j * 16 + l16;
            const int head = oc >> 7;
            const int d    = oc & (HDIM - 1);
#pragma unroll
            for (int r = 0; r < 8; ++r) {
                const int row = m0 + wm * 64 + i * 16 + r + ksel * 8;
                const int b   = row >> 12;
                const int s   = row & (SEQ - 1);
                const size_t off = ((size_t)(b * NHEADS + head) * SEQ + s) * HDIM + d;
                outW[off] = acc[i][j][r];
            }
        }
    }
}

// ---------------------------------------------------------------------------
// Fallback (round-1): single kernel with in-loop conversion, 128x128 tiles.
// Used only if the workspace is too small for the hi/lo arrays.
// ---------------------------------------------------------------------------
#define FLDA 40

__device__ __forceinline__ void fb_load_stage(const float* __restrict__ base, int kb,
                                              int tid, float4* r) {
#pragma unroll
    for (int i = 0; i < 4; ++i) {
        int f = tid + i * 256, row = f >> 3, c4 = f & 7;
        r[i] = *(const float4*)(base + (size_t)row * HIDDEN + kb + c4 * 4);
    }
}
__device__ __forceinline__ void fb_store_stage(unsigned short* hiBuf, unsigned short* loBuf,
                                               int tid, const float4* r) {
#pragma unroll
    for (int i = 0; i < 4; ++i) {
        int f = tid + i * 256, row = f >> 3, c4 = f & 7;
        const float* v = (const float*)&r[i];
        unsigned h[4], l[4];
#pragma unroll
        for (int e = 0; e < 4; ++e) {
            h[e] = bf16rne(v[e]);
            l[e] = bf16rne(v[e] - bf16tof(h[e]));
        }
        unsigned off = (unsigned)row * FLDA + (unsigned)c4 * 4u;
        u32x2 hv = { h[0] | (h[1] << 16), h[2] | (h[3] << 16) };
        u32x2 lv = { l[0] | (l[1] << 16), l[2] | (l[3] << 16) };
        *(u32x2*)(hiBuf + off) = hv;
        *(u32x2*)(loBuf + off) = lv;
    }
}

__global__ __launch_bounds__(256)
void fused_qkv_bf16x3_fb(const float* __restrict__ A,
                         const float* __restrict__ Wq,
                         const float* __restrict__ Wk,
                         const float* __restrict__ Wv,
                         float* __restrict__ out) {
    __shared__ __align__(16) unsigned short sAhi[128 * FLDA];
    __shared__ __align__(16) unsigned short sAlo[128 * FLDA];
    __shared__ __align__(16) unsigned short sBhi[128 * FLDA];
    __shared__ __align__(16) unsigned short sBlo[128 * FLDA];

    const int tid = threadIdx.x, lane = tid & 31, wave = tid >> 5;
    const int wm = wave & 1, wn = wave >> 1, l16 = lane & 15, ksel = lane >> 4;
    const int m0 = blockIdx.x * 128, n0 = blockIdx.y * 128;
    const int which = n0 >> 11, o0 = n0 & (HIDDEN - 1);
    const float* Wsel = (which == 0) ? Wq : (which == 1 ? Wk : Wv);
    const float* Abase = A + (size_t)m0 * HIDDEN;
    const float* Bbase = Wsel + (size_t)o0 * HIDDEN;

    f32x8 acc[4][2] = {};
    float4 aCur[4], bCur[4], aNxt[4], bNxt[4];
    fb_load_stage(Abase, 0, tid, aCur);
    fb_load_stage(Bbase, 0, tid, bCur);

    for (int kb = 0; kb < HIDDEN; kb += 32) {
        __syncthreads();
        fb_store_stage(sAhi, sAlo, tid, aCur);
        fb_store_stage(sBhi, sBlo, tid, bCur);
        __syncthreads();
        const bool more = (kb + 32) < HIDDEN;
        if (more) {
            fb_load_stage(Abase, kb + 32, tid, aNxt);
            fb_load_stage(Bbase, kb + 32, tid, bNxt);
        }
        bf16x16 bh[2], bl[2];
#pragma unroll
        for (int j = 0; j < 2; ++j) {
            unsigned off = (unsigned)(wn * 32 + j * 16 + l16) * FLDA + (unsigned)ksel * 8u;
            bh[j] = load_frag(sBhi + off, 16);
            bl[j] = load_frag(sBlo + off, 16);
        }
#pragma unroll
        for (int i = 0; i < 4; ++i) {
            unsigned off = (unsigned)(wm * 64 + i * 16 + l16) * FLDA + (unsigned)ksel * 8u;
            bf16x16 ah = load_frag(sAhi + off, 16);
            bf16x16 al = load_frag(sAlo + off, 16);
#pragma unroll
            for (int j = 0; j < 2; ++j) {
                acc[i][j] = __builtin_amdgcn_wmma_f32_16x16x32_bf16(
                    false, ah, false, bh[j], (short)0, acc[i][j], false, false);
                acc[i][j] = __builtin_amdgcn_wmma_f32_16x16x32_bf16(
                    false, ah, false, bl[j], (short)0, acc[i][j], false, false);
                acc[i][j] = __builtin_amdgcn_wmma_f32_16x16x32_bf16(
                    false, al, false, bh[j], (short)0, acc[i][j], false, false);
            }
        }
        if (more) {
#pragma unroll
            for (int i = 0; i < 4; ++i) { aCur[i] = aNxt[i]; bCur[i] = bNxt[i]; }
        }
    }

    float* outW = out + (size_t)which * OUT_PER;
#pragma unroll
    for (int i = 0; i < 4; ++i)
#pragma unroll
        for (int j = 0; j < 2; ++j) {
            const int oc = o0 + wn * 32 + j * 16 + l16;
            const int head = oc >> 7, d = oc & (HDIM - 1);
#pragma unroll
            for (int r = 0; r < 8; ++r) {
                const int row = m0 + wm * 64 + i * 16 + r + ksel * 8;
                const int b = row >> 12, s = row & (SEQ - 1);
                outW[((size_t)(b * NHEADS + head) * SEQ + s) * HDIM + d] = acc[i][j][r];
            }
        }
}

// ---------------------------------------------------------------------------
extern "C" void kernel_launch(void* const* d_in, const int* in_sizes, int n_in,
                              void* d_out, int out_size, void* d_ws, size_t ws_size,
                              hipStream_t stream) {
    (void)in_sizes; (void)n_in; (void)out_size;
    const float* hs = (const float*)d_in[0];
    const float* wq = (const float*)d_in[1];
    const float* wk = (const float*)d_in[2];
    const float* wv = (const float*)d_in[3];
    float* out = (float*)d_out;

    const size_t A_ELEMS = (size_t)MTOT * HIDDEN;       // 16.78M
    const size_t W_ELEMS = (size_t)HIDDEN * HIDDEN;     // 4.19M per matrix
    const size_t AHI_B = A_ELEMS * 2;                   // bytes
    const size_t WHI_B = 3 * W_ELEMS * 2;
    const size_t NEED  = 2 * AHI_B + 2 * WHI_B;         // 117,440,512

    if (ws_size >= NEED) {
        unsigned short* aHi = (unsigned short*)d_ws;
        unsigned short* aLo = aHi + A_ELEMS;
        unsigned short* wHi = aLo + A_ELEMS;            // [6144][2048] concat q|k|v
        unsigned short* wLo = wHi + 3 * W_ELEMS;

        int nA8 = (int)(A_ELEMS / 8);
        int nW8 = (int)(W_ELEMS / 8);
        split_bf16_kernel<<<(nA8 + 255) / 256, 256, 0, stream>>>(hs, aHi, aLo, nA8);
        split_bf16_kernel<<<(nW8 + 255) / 256, 256, 0, stream>>>(wq, wHi, wLo, nW8);
        split_bf16_kernel<<<(nW8 + 255) / 256, 256, 0, stream>>>(
            wk, wHi + W_ELEMS, wLo + W_ELEMS, nW8);
        split_bf16_kernel<<<(nW8 + 255) / 256, 256, 0, stream>>>(
            wv, wHi + 2 * W_ELEMS, wLo + 2 * W_ELEMS, nW8);

        dim3 grid(MTOT / BM, NTOT / BN);   // 32 x 48
        fused_qkv_gemm_async<<<grid, 512, 2 * L_BUF, stream>>>(aHi, aLo, wHi, wLo, out);
    } else {
        dim3 grid(MTOT / 128, NTOT / 128); // 64 x 48
        fused_qkv_bf16x3_fb<<<grid, 256, 0, stream>>>(hs, wq, wk, wv, out);
    }
}
